// PTPDNet_SumOfTwo_34411277975826
// MI455X (gfx1250) — matrix operand order repaired and checked by reading, not compiled, not run on hardware
//
#include <hip/hip_runtime.h>

typedef __attribute__((ext_vector_type(16))) __bf16 v16bf;
typedef __attribute__((ext_vector_type(8)))  float  v8f;
typedef __attribute__((ext_vector_type(4)))  int    v4i;

union Frag16 { v16bf v; v4i q[2]; };

#define EPSQ 1e-6f

// ---- shared memory layout (bytes) ----
#define OFF_WQ0   0           // 128x32  bf16 (8192)   Wq0 padded K16->32
#define OFF_WQ1   8192        // 128x128 bf16 (32768)
#define OFF_WQ2   40960       // 144x128 bf16 (36864)  Wq2 padded N136->144
#define OFF_WA0   77824       // 64x32   bf16 (4096)   Wa0 padded K16->32
#define OFF_WA1   81920       // 64x64   bf16 (8192)
#define OFF_BQ0   90112       // 128 f32
#define OFF_BQ1   90624       // 128 f32
#define OFF_BQ2   91136       // 144 f32
#define OFF_STAGE 91712
// per-wave staging: X bf16 16x32 (1024) | Act bf16 16x128 (4096) |
//                   Q f32 16x144 (9216) | A f32 16x64 (4096)
#define STG_X     0
#define STG_ACT   1024
#define STG_Q     5120
#define STG_A     14336
#define STAGE_PER_WAVE 18432
#define SHMEM_TOTAL (OFF_STAGE + 8 * STAGE_PER_WAVE)   // 239168 bytes

__device__ __forceinline__ unsigned short f2bf(float f) {
  unsigned int u = __float_as_uint(f);
  u += 0x7FFFu + ((u >> 16) & 1u);       // round-to-nearest-even
  return (unsigned short)(u >> 16);
}

// Branchless tanh via hardware transcendentals:
//   tanh(x) = 1 - 2 / (e^{2x} + 1)
// x->+inf: e=inf -> rcp=0 -> 1;  x->-inf: e=0 -> rcp(1)=1 -> -1.
// Lowers to v_mul + v_exp_f32 + v_add + v_rcp_f32 + v_fma (no EXEC branches).
__device__ __forceinline__ float fast_tanh(float x) {
  const float e = __expf(x + x);
  const float r = __builtin_amdgcn_rcpf(e + 1.0f);
  return fmaf(-2.0f, r, 1.0f);
}

__device__ __forceinline__ void lds_fence() {
  // order prior DS stores vs subsequent DS loads within this wave;
  // also a full compiler memory barrier.
  asm volatile("s_wait_dscnt 0" ::: "memory");
}

// One dense layer on a 16-row batch tile. actIn: bf16 [16][skIn] in LDS.
// W: bf16 [NTILES*16][skW] in LDS (row-major [N][K]). bias: f32 or null.
// A fragment (16x32 bf16): lane m=l&15, hi=l>>4: elems 0-7 -> k = ks*32+8*hi+e,
// elems 8-15 -> k = ks*32+16+8*hi+(e-8).  (ISA 7.12.2 16-bit A layout)
// B fragment (32x16 bf16): lane n=l&15, hi=l>>4: same k mapping as A.
// C/D (16x16 f32): lane n=l&15; vgpr r -> row m = r + 8*(l>>4).
template<int KSTEPS, int NTILES, bool DO_TANH, bool STORE_BF, bool STORE_F32>
__device__ __forceinline__ void layer(const unsigned short* actIn, int skIn,
                                      const unsigned short* W, int skW,
                                      const float* bias,
                                      unsigned short* actOut, int skOut,
                                      float* f32Out, int skF,
                                      int lane) {
  const int n  = lane & 15;
  const int hi = lane >> 4;
  v8f acc[NTILES];
#pragma unroll
  for (int nt = 0; nt < NTILES; ++nt) {
    const float bv = bias ? bias[nt * 16 + n] : 0.0f;
#pragma unroll
    for (int r = 0; r < 8; ++r) acc[nt][r] = bv;
  }
  lds_fence();
#pragma unroll
  for (int ks = 0; ks < KSTEPS; ++ks) {
    Frag16 a;
    const unsigned short* ap = actIn + n * skIn + ks * 32 + 8 * hi;
    a.q[0] = *(const v4i*)(ap);
    a.q[1] = *(const v4i*)(ap + 16);
#pragma unroll
    for (int nt = 0; nt < NTILES; ++nt) {
      Frag16 b;
      const unsigned short* wp = W + (nt * 16 + n) * skW + ks * 32 + 8 * hi;
      b.q[0] = *(const v4i*)(wp);
      b.q[1] = *(const v4i*)(wp + 16);
      acc[nt] = __builtin_amdgcn_wmma_f32_16x16x32_bf16(
          false, a.v, false, b.v, (short)0, acc[nt], false, false);
    }
  }
#pragma unroll
  for (int nt = 0; nt < NTILES; ++nt) {
#pragma unroll
    for (int r = 0; r < 8; ++r) {
      float v = acc[nt][r];
      if (DO_TANH) v = fast_tanh(v);
      const int m = r + 8 * hi;
      const int col = nt * 16 + n;
      if (STORE_BF)  actOut[m * skOut + col] = f2bf(v);
      if (STORE_F32) f32Out[m * skF + col] = v;
    }
  }
}

__global__ void __launch_bounds__(256)
ptpd_sum_of_two_kernel(const float* __restrict__ points,
                       const float* __restrict__ Wq0, const float* __restrict__ bq0,
                       const float* __restrict__ Wq1, const float* __restrict__ bq1,
                       const float* __restrict__ Wq2, const float* __restrict__ bq2,
                       const float* __restrict__ Wa0, const float* __restrict__ Wa1,
                       float* __restrict__ out, int nTiles) {
  extern __shared__ char smem[];
  unsigned short* sWq0 = (unsigned short*)(smem + OFF_WQ0);
  unsigned short* sWq1 = (unsigned short*)(smem + OFF_WQ1);
  unsigned short* sWq2 = (unsigned short*)(smem + OFF_WQ2);
  unsigned short* sWa0 = (unsigned short*)(smem + OFF_WA0);
  unsigned short* sWa1 = (unsigned short*)(smem + OFF_WA1);
  float* sbq0 = (float*)(smem + OFF_BQ0);
  float* sbq1 = (float*)(smem + OFF_BQ1);
  float* sbq2 = (float*)(smem + OFF_BQ2);

  const int tid = threadIdx.x;
  const int nth = blockDim.x;

  // ---- cooperative weight preload: f32 -> bf16, zero-padded ----
  for (int i = tid; i < 128 * 32; i += nth) {
    const int r = i >> 5, k = i & 31;
    sWq0[i] = (k < 16) ? f2bf(Wq0[r * 16 + k]) : (unsigned short)0;
  }
  for (int i = tid; i < 128 * 128; i += nth) sWq1[i] = f2bf(Wq1[i]);
  for (int i = tid; i < 144 * 128; i += nth) {
    const int r = i >> 7;
    sWq2[i] = (r < 136) ? f2bf(Wq2[i]) : (unsigned short)0;
  }
  for (int i = tid; i < 64 * 32; i += nth) {
    const int r = i >> 5, k = i & 31;
    sWa0[i] = (k < 16) ? f2bf(Wa0[r * 16 + k]) : (unsigned short)0;
  }
  for (int i = tid; i < 64 * 64; i += nth) sWa1[i] = f2bf(Wa1[i]);
  for (int i = tid; i < 128; i += nth) { sbq0[i] = bq0[i]; sbq1[i] = bq1[i]; }
  for (int i = tid; i < 144; i += nth) sbq2[i] = (i < 136) ? bq2[i] : 0.0f;
  __syncthreads();

  const int wave = tid >> 5;
  const int lane = tid & 31;
  char* stage = smem + OFF_STAGE + wave * STAGE_PER_WAVE;
  unsigned short* sX   = (unsigned short*)(stage + STG_X);    // 16x32 bf16
  unsigned short* sAct = (unsigned short*)(stage + STG_ACT);  // 16x128 bf16
  float* sQ = (float*)(stage + STG_Q);                        // 16x144 f32
  float* sA = (float*)(stage + STG_A);                        // 16x64 f32

  const int gwave  = blockIdx.x * (nth >> 5) + wave;
  const int nwaves = gridDim.x * (nth >> 5);

  for (int tile = gwave; tile < nTiles; tile += nwaves) {
    const int rowBase = tile << 4;

    // ---- stage X: f32 global -> bf16 [16][32] LDS, zero pad k=16..31 ----
    {
      const int r  = lane & 15;
      const int cb = (lane >> 4) * 8;
      const float* px = points + (size_t)(rowBase + r) * 16 + cb;
      const float4 p0 = *(const float4*)px;
      const float4 p1 = *(const float4*)(px + 4);
      unsigned short* dst = sX + r * 32 + cb;
      dst[0] = f2bf(p0.x); dst[1] = f2bf(p0.y);
      dst[2] = f2bf(p0.z); dst[3] = f2bf(p0.w);
      dst[4] = f2bf(p1.x); dst[5] = f2bf(p1.y);
      dst[6] = f2bf(p1.z); dst[7] = f2bf(p1.w);
      unsigned short* dz = sX + r * 32 + 16 + cb;
#pragma unroll
      for (int j = 0; j < 8; ++j) dz[j] = 0;
    }

    // ---- quad tower: 16->128->128->136(144) ----
    layer<1, 8, true,  true,  false>(sX, 32,  sWq0, 32,  sbq0, sAct, 128, nullptr, 0, lane);
    layer<4, 8, true,  true,  false>(sAct, 128, sWq1, 128, sbq1, sAct, 128, nullptr, 0, lane);
    layer<4, 9, false, false, true >(sAct, 128, sWq2, 128, sbq2, nullptr, 0, sQ, 144, lane);

    // ---- add tower: 16->64->64 ----
    layer<1, 4, true,  true,  false>(sX, 32,  sWa0, 32, nullptr, sAct, 64, nullptr, 0, lane);
    layer<2, 4, true,  false, true >(sAct, 64, sWa1, 64, nullptr, nullptr, 0, sA, 64, lane);

    lds_fence();

    // ---- per-sample reduction: 2 lanes per sample (h = column/feature half) ----
    {
      const int s = lane & 15;
      const int h = lane >> 4;
      const float* xr = points + (size_t)(rowBase + s) * 16;
      float x[16];
#pragma unroll
      for (int i = 0; i < 16; ++i) x[i] = xr[i];

      const float* Ls = sQ + s * 144;
      float vq = 0.0f;
#pragma unroll
      for (int j8 = 0; j8 < 8; ++j8) {
        const int j = h * 8 + j8;              // column of L (0..15)
        float z = 0.0f;
#pragma unroll
        for (int i = 0; i < 16; ++i) {         // z_j = sum_{i>=j} L[i][j]*x[i]
          const float Lij = Ls[(i * (i + 1)) / 2 + j];
          z = fmaf((i >= j) ? Lij : 0.0f, x[i], z);
        }
        vq = fmaf(z, z, vq);
      }
      float vx = 0.0f;
#pragma unroll
      for (int i8 = 0; i8 < 8; ++i8) {
        const float xi = x[h * 8 + i8];
        vx = fmaf(xi, xi, vx);
      }
      float va = 0.0f;
      const float* As = sA + s * 64 + h * 32;
#pragma unroll
      for (int k = 0; k < 32; ++k) {
        const float a = As[k];
        va = fmaf(a, a, va);
      }
      float tot = vq + EPSQ * vx + va;
      tot += __shfl_xor(tot, 16, 32);
      if (h == 0) out[rowBase + s] = tot;
    }
  }
}

extern "C" void kernel_launch(void* const* d_in, const int* in_sizes, int n_in,
                              void* d_out, int out_size, void* d_ws, size_t ws_size,
                              hipStream_t stream) {
  (void)n_in; (void)d_ws; (void)ws_size; (void)out_size;
  const float* points = (const float*)d_in[0];
  const float* Wq0 = (const float*)d_in[1];
  const float* bq0 = (const float*)d_in[2];
  const float* Wq1 = (const float*)d_in[3];
  const float* bq1 = (const float*)d_in[4];
  const float* Wq2 = (const float*)d_in[5];
  const float* bq2 = (const float*)d_in[6];
  const float* Wa0 = (const float*)d_in[7];
  const float* Wa1 = (const float*)d_in[8];
  float* out = (float*)d_out;

  const int B = in_sizes[0] / 16;      // number of samples
  const int nTiles = B / 16;           // 16 samples per wave-tile
  const dim3 grid(256), block(256);    // 8 wave32 per block
  ptpd_sum_of_two_kernel<<<grid, block, SHMEM_TOTAL, stream>>>(
      points, Wq0, bq0, Wq1, bq1, Wq2, bq2, Wa0, Wa1, out, nTiles);
}
